// Attention_9663676416195
// MI455X (gfx1250) — matrix operand order repaired and checked by reading, not compiled, not run on hardware
//
#include <hip/hip_runtime.h>

// ---------------------------------------------------------------------------
// CDNA5 (gfx1250) attention block: bf16 WMMA + async-to-LDS staged GEMMs.
// ---------------------------------------------------------------------------

typedef __attribute__((ext_vector_type(8)))  __bf16 bhalf8;
typedef __attribute__((ext_vector_type(16))) __bf16 bhalf16;
typedef __attribute__((ext_vector_type(8)))  float  v8f;

#define DIM  4096
#define NH   32
#define NKV  8
#define HD   128
#define BB   2
#define SS   2048
#define ROWS (BB * SS)       /* 4096 token rows */
#define KVD  (NKV * HD)      /* 1024 */
#define LN_EPS 1e-5f

union BF16x16 { bhalf16 v; bhalf8 h[2]; };

static __device__ __forceinline__ bhalf16 mk16(bhalf8 lo, bhalf8 hi) {
  BF16x16 u; u.h[0] = lo; u.h[1] = hi; return u.v;
}

static __device__ __forceinline__ v8f wmma_bf16(bhalf16 a, bhalf16 b, v8f c) {
  // D = A(16x32) * B(32x16) + C, fp32 accumulate
  return __builtin_amdgcn_wmma_f32_16x16x32_bf16(false, a, false, b, (short)0, c,
                                                 false, false);
}

// --- async global->LDS copy (CDNA5 GLOBAL_LOAD_ASYNC_TO_LDS, ASYNCcnt) ------
typedef __attribute__((address_space(3))) void lds_void;

static __device__ __forceinline__ unsigned lds_offset(void* p) {
  // generic -> LDS addrspace, then ptrtoint via 64-bit (host pass safe; on
  // device the AS3 pointer is 32-bit so truncation is exact).
  return (unsigned)(unsigned long long)(lds_void*)p;
}

static __device__ __forceinline__ void async_copy_b128(const void* gptr, void* lptr) {
  unsigned l = lds_offset(lptr);
  asm volatile("global_load_async_to_lds_b128 %0, %1, off"
               :: "v"(l), "v"(gptr) : "memory");
}

static __device__ __forceinline__ void wait_async_le10() {
  asm volatile("s_wait_asynccnt 10" ::: "memory");
}
static __device__ __forceinline__ void wait_async_0() {
  asm volatile("s_wait_asynccnt 0" ::: "memory");
}

// ---------------------------------------------------------------------------
// fp32 -> bf16 elementwise convert
// ---------------------------------------------------------------------------
__global__ void cvt_kernel(const float* __restrict__ in, __bf16* __restrict__ out,
                           int n) {
  int i = blockIdx.x * blockDim.x + threadIdx.x;
  int stride = gridDim.x * blockDim.x;
  for (; i < n; i += stride) out[i] = (__bf16)in[i];
}

// ---------------------------------------------------------------------------
// Tiled GEMM: C[M,N] = A[M,K] * W[N,K]^T   (bf16 in, fp32 or bf16 out)
// Block = 256 threads (8 waves). Block tile 64(M) x 256(N), K-step 64.
// Double-buffered LDS staging via global_load_async_to_lds_b128
// (per thread per stage: 2 A + 8 W b128 copies -> s_wait_asynccnt 10/0).
// Each wave computes a 32x64 sub-tile: per k32 step 2 A-frags + 4 B-frags
// feed 8x v_wmma_f32_16x16x32_bf16 (8 accumulators, 64 acc VGPRs).
// Fragment layouts (ISA 7.12.2):
//   A-frag 16x32: lane m=lane&15; lo lanes K 0-7 & 16-23, hi lanes K 8-15 & 24-31
//   B-frag 32x16: lane n=lane&15; lo lanes K 0-7 & 16-23, hi lanes K 8-15 & 24-31
// ---------------------------------------------------------------------------
#define BM  64
#define BN  256
#define BK  64

__global__ void gemm_tiled_kernel(const __bf16* __restrict__ A,
                                  const __bf16* __restrict__ W,
                                  float* __restrict__ outF, __bf16* __restrict__ outB,
                                  int M, int N, int K) {
  __shared__ __bf16 sA[2][BM * BK];   // row-major, row stride BK (8 KB / buf)
  __shared__ __bf16 sW[2][BN * BK];   // row-major, row stride BK (32 KB / buf)

  int tid = threadIdx.x;
  int nbt = N / BN;
  int mt = blockIdx.x / nbt, nt = blockIdx.x % nbt;
  const __bf16* Ag = A + (size_t)mt * BM * K;
  const __bf16* Wg = W + (size_t)nt * BN * K;

  int arow = tid >> 2, aseg = (tid & 3) * 2;  // A: 64 rows x 8 b128 segs, 2/thread

  auto issue = [&](int kb, int buf) {
    int k0 = kb * BK;
    async_copy_b128(Ag + (size_t)arow * K + k0 + aseg * 8,
                    &sA[buf][arow * BK + aseg * 8]);
    async_copy_b128(Ag + (size_t)arow * K + k0 + (aseg + 1) * 8,
                    &sA[buf][arow * BK + (aseg + 1) * 8]);
#pragma unroll
    for (int s2 = 0; s2 < 8; ++s2)
      async_copy_b128(Wg + (size_t)tid * K + k0 + s2 * 8,
                      &sW[buf][tid * BK + s2 * 8]);
  };

  int wid = tid >> 5, lane = tid & 31, l = lane & 15, hi = lane >> 4;
  int wm = (wid & 1) * 32;    // wave M offset within block tile (2 M-tiles)
  int wn = (wid >> 1) * 64;   // wave N offset within block tile

  v8f acc[2][4] = {};
  const int KB = K / BK;

  issue(0, 0);
  for (int kb = 0; kb < KB; ++kb) {
    int buf = kb & 1;
    if (kb + 1 < KB) {
      issue(kb + 1, buf ^ 1);   // prefetch next stage (10 more in flight)
      wait_async_le10();        // oldest 10 (= stage kb) complete, in-order
    } else {
      wait_async_0();
    }
    __syncthreads();            // publish tile across all 8 waves

    const __bf16* a0 = &sA[buf][(wm + l) * BK + hi * 8];        // rows wm..+15
    const __bf16* a1 = a0 + 16 * BK;                            // rows wm+16..+31
    const __bf16* w0 = &sW[buf][(wn + l) * BK + hi * 8];
#pragma unroll
    for (int k = 0; k < BK; k += 32) {
      bhalf16 A0 = mk16(*(const bhalf8*)(a0 + k), *(const bhalf8*)(a0 + k + 16));
      bhalf16 A1 = mk16(*(const bhalf8*)(a1 + k), *(const bhalf8*)(a1 + k + 16));
#pragma unroll
      for (int j = 0; j < 4; ++j) {
        const __bf16* wj = w0 + (size_t)(16 * j) * BK + k;
        bhalf16 Bj = mk16(*(const bhalf8*)wj, *(const bhalf8*)(wj + 16));
        acc[0][j] = wmma_bf16(A0, Bj, acc[0][j]);
        acc[1][j] = wmma_bf16(A1, Bj, acc[1][j]);
      }
    }
    __syncthreads();            // all waves done reading buf before overwrite
  }

  // C layout: VGPR r, lanes 0-15 -> M=r, N=lane; lanes 16-31 -> M=r+8.
#pragma unroll
  for (int t = 0; t < 2; ++t) {
#pragma unroll
    for (int r = 0; r < 8; ++r) {
      int row = mt * BM + wm + t * 16 + r + hi * 8;
      size_t base = (size_t)row * N + nt * BN + wn + l;
      if (outF) {
#pragma unroll
        for (int j = 0; j < 4; ++j) outF[base + 16 * j] = acc[t][j][r];
      } else {
#pragma unroll
        for (int j = 0; j < 4; ++j) outB[base + 16 * j] = (__bf16)acc[t][j][r];
      }
    }
  }
}

// ---------------------------------------------------------------------------
// In-place LayerNorm over full E row + RoPE per head (bf16 in/out).
// One 256-thread block per token row.
// ---------------------------------------------------------------------------
__global__ void lnrope_kernel(__bf16* __restrict__ t, const float* __restrict__ w,
                              const float* __restrict__ bias,
                              const float* __restrict__ fc, int E) {
  int row = blockIdx.x;
  int s = row & (SS - 1);
  size_t base = (size_t)row * E;
  int tid = threadIdx.x;

  float sum = 0.f, sq = 0.f;
  for (int i = tid; i < E; i += blockDim.x) {
    float v = (float)t[base + i];
    sum += v; sq += v * v;
  }
#pragma unroll
  for (int off = 16; off >= 1; off >>= 1) {
    sum += __shfl_xor(sum, off);
    sq  += __shfl_xor(sq, off);
  }
  __shared__ float ps[8], pq[8];
  int wid = tid >> 5;
  if ((tid & 31) == 0) { ps[wid] = sum; pq[wid] = sq; }
  __syncthreads();
  float ts = 0.f, tq = 0.f;
  int nw = blockDim.x >> 5;
  for (int i = 0; i < nw; ++i) { ts += ps[i]; tq += pq[i]; }
  float mu = ts / (float)E;
  float var = tq / (float)E - mu * mu;
  float rstd = rsqrtf(var + LN_EPS);
  __syncthreads();

  for (int idx = tid * 2; idx < E; idx += blockDim.x * 2) {
    float y0 = ((float)t[base + idx]     - mu) * rstd * w[idx]     + bias[idx];
    float y1 = ((float)t[base + idx + 1] - mu) * rstd * w[idx + 1] + bias[idx + 1];
    int i = (idx & (HD - 1)) >> 1;            // pair index within head
    float c  = fc[(s * (HD / 2) + i) * 2 + 0];
    float sn = fc[(s * (HD / 2) + i) * 2 + 1];
    t[base + idx]     = (__bf16)(y0 * c - y1 * sn);
    t[base + idx + 1] = (__bf16)(y0 * sn + y1 * c);
  }
}

// ---------------------------------------------------------------------------
// V transpose: [b, s, kv*HD] -> [b, kv, hd, s]  (bf16)
// ---------------------------------------------------------------------------
__global__ void vtrans_kernel(const __bf16* __restrict__ v, __bf16* __restrict__ vt) {
  int i = blockIdx.x * blockDim.x + threadIdx.x;
  const int total = ROWS * KVD;
  if (i >= total) return;
  int hd   = i & (KVD - 1);
  int srow = i >> 10;            // b*SS + s
  int b    = srow >> 11;         // SS = 2048
  int s    = srow & (SS - 1);
  int kvh  = hd >> 7;
  int d    = hd & (HD - 1);
  vt[(((size_t)(b * NKV + kvh)) * HD + d) * SS + s] = v[i];
}

// ---------------------------------------------------------------------------
// GQA flash attention, online softmax. One wave per (b, h, 16-query tile).
// Scores computed transposed: C = K_tile(16 x hd) * Q^T(hd x 16) so each lane
// owns one query column; softmax over keys = 16 in-lane vals + shfl_xor(16).
// P goes through LDS (b128 packed) to be re-read in A-frag layout for PV.
// ---------------------------------------------------------------------------
__global__ void attn_kernel(const __bf16* __restrict__ q, const __bf16* __restrict__ k,
                            const __bf16* __restrict__ vt, __bf16* __restrict__ o) {
  __shared__ __bf16 plds[4][16 * 40];   // per-wave 16 rows x 32 keys (stride 40)

  int gw = blockIdx.x * (blockDim.x >> 5) + (threadIdx.x >> 5);
  int qt = gw & 127;              // SS/16 = 128 query tiles
  int h  = (gw >> 7) & 31;
  int b  = gw >> 12;
  int kv = h >> 2;                // R = NH/NKV = 4
  int lane = threadIdx.x & 31, l = lane & 15, hi = lane >> 4;
  __bf16* pl = plds[threadIdx.x >> 5];
  const float scale = 0.08838834764831845f;   // 1/sqrt(128)

  // Q as B-frags: column n = query l, hd values hi*8..+7 and 16+hi*8..+7 per lane.
  const __bf16* qb_ = q + ((size_t)(b * SS + qt * 16 + l)) * DIM + h * HD + hi * 8;
  bhalf16 qf[4];
#pragma unroll
  for (int c = 0; c < 4; ++c)
    qf[c] = mk16(*(const bhalf8*)(qb_ + c * 32), *(const bhalf8*)(qb_ + c * 32 + 16));

  const __bf16* kb_ = k + ((size_t)b * SS) * KVD + kv * HD + (size_t)l * KVD + hi * 8;
  const __bf16* vb_ = vt + ((size_t)(b * NKV + kv)) * HD * SS + (size_t)l * SS + hi * 8;

  v8f o_[8] = {};
  float m_run = -3.0e38f, l_run = 0.f;

  for (int kt = 0; kt < SS / 32; ++kt) {
    const __bf16* kp = kb_ + (size_t)kt * 32 * KVD;
    v8f c0 = {}, c1 = {};
#pragma unroll
    for (int c = 0; c < 4; ++c) {
      bhalf16 a0 = mk16(*(const bhalf8*)(kp + c * 32),
                        *(const bhalf8*)(kp + c * 32 + 16));
      bhalf16 a1 = mk16(*(const bhalf8*)(kp + (size_t)16 * KVD + c * 32),
                        *(const bhalf8*)(kp + (size_t)16 * KVD + c * 32 + 16));
      c0 = wmma_bf16(a0, qf[c], c0);   // keys 0-15 of group
      c1 = wmma_bf16(a1, qf[c], c1);   // keys 16-31 of group
    }

    float tm = -3.0e38f;
#pragma unroll
    for (int r = 0; r < 8; ++r) {
      c0[r] *= scale; c1[r] *= scale;
      tm = fmaxf(tm, fmaxf(c0[r], c1[r]));
    }
    tm = fmaxf(tm, __shfl_xor(tm, 16));
    float mn  = fmaxf(m_run, tm);
    float fc_ = __expf(m_run - mn);
    float rs = 0.f;
    bhalf8 p0, p1;
#pragma unroll
    for (int r = 0; r < 8; ++r) {
      float e0 = __expf(c0[r] - mn);
      float e1 = __expf(c1[r] - mn);
      rs += e0 + e1;
      p0[r] = (__bf16)e0;
      p1[r] = (__bf16)e1;
    }
    rs += __shfl_xor(rs, 16);
    l_run = l_run * fc_ + rs;
    m_run = mn;

    // rescale running O: factor indexed by query M = r + hi*8 (source lane = q)
    float fr[8];
#pragma unroll
    for (int r = 0; r < 8; ++r) fr[r] = __shfl(fc_, r + hi * 8);
#pragma unroll
    for (int nc = 0; nc < 8; ++nc) {
#pragma unroll
      for (int r = 0; r < 8; ++r) o_[nc][r] *= fr[r];
    }

    // stage P (query-major, 32 keys, stride 40) for A-frag reread
    *(bhalf8*)&pl[l * 40 + hi * 8]      = p0;   // keys hi*8 .. +7
    *(bhalf8*)&pl[l * 40 + 16 + hi * 8] = p1;   // keys 16+hi*8 .. +7
    asm volatile("s_wait_dscnt 0" ::: "memory");
    __builtin_amdgcn_wave_barrier();
    bhalf16 pa = mk16(*(const bhalf8*)&pl[l * 40 + hi * 8],
                      *(const bhalf8*)&pl[l * 40 + hi * 8 + 16]);

    // O += P(16q x 32keys) * V(32keys x 16hd), 8 hd chunks
#pragma unroll
    for (int nc = 0; nc < 8; ++nc) {
      const __bf16* vp = vb_ + (size_t)(nc * 16) * SS + kt * 32;
      bhalf16 bv = mk16(*(const bhalf8*)vp, *(const bhalf8*)(vp + 16));
      o_[nc] = wmma_bf16(pa, bv, o_[nc]);
    }
  }

  float li = 1.0f / l_run;
  float lr[8];
#pragma unroll
  for (int r = 0; r < 8; ++r) lr[r] = __shfl(li, r + hi * 8);

  __bf16* ob_ = o + ((size_t)(b * SS + qt * 16)) * DIM + h * HD;
#pragma unroll
  for (int nc = 0; nc < 8; ++nc) {
#pragma unroll
    for (int r = 0; r < 8; ++r)
      ob_[(size_t)(r + hi * 8) * DIM + nc * 16 + l] = (__bf16)(o_[nc][r] * lr[r]);
  }
}

// ---------------------------------------------------------------------------
// Host-side orchestration
// ---------------------------------------------------------------------------
extern "C" void kernel_launch(void* const* d_in, const int* in_sizes, int n_in,
                              void* d_out, int out_size, void* d_ws, size_t ws_size,
                              hipStream_t stream) {
  (void)in_sizes; (void)n_in; (void)out_size; (void)ws_size;
  const float* x  = (const float*)d_in[0];
  const float* fc = (const float*)d_in[1];
  const float* wq = (const float*)d_in[2];
  const float* wk = (const float*)d_in[3];
  const float* wv = (const float*)d_in[4];
  const float* wo = (const float*)d_in[5];
  const float* qw = (const float*)d_in[6];
  const float* qb = (const float*)d_in[7];
  const float* kw = (const float*)d_in[8];
  const float* kb = (const float*)d_in[9];

  char* ws = (char*)d_ws;
  const size_t MB = 1024 * 1024;
  __bf16* xb   = (__bf16*)(ws);              // 32 MB  x bf16
  __bf16* wqb  = (__bf16*)(ws + 32 * MB);    // 32 MB
  __bf16* wkb  = (__bf16*)(ws + 64 * MB);    //  8 MB
  __bf16* wvb  = (__bf16*)(ws + 72 * MB);    //  8 MB
  __bf16* wob  = (__bf16*)(ws + 80 * MB);    // 32 MB
  __bf16* qbuf = (__bf16*)(ws + 112 * MB);   // 32 MB  q proj (ln+rope in place)
  __bf16* kbuf = (__bf16*)(ws + 144 * MB);   //  8 MB
  __bf16* vbuf = (__bf16*)(ws + 152 * MB);   //  8 MB
  __bf16* vtb  = (__bf16*)(ws + 160 * MB);   //  8 MB  V transposed
  __bf16* ab   = (__bf16*)(ws + 168 * MB);   // 32 MB  attention out

  cvt_kernel<<<4096, 256, 0, stream>>>(x,  xb,  ROWS * DIM);
  cvt_kernel<<<4096, 256, 0, stream>>>(wq, wqb, DIM * DIM);
  cvt_kernel<<<4096, 256, 0, stream>>>(wk, wkb, KVD * DIM);
  cvt_kernel<<<4096, 256, 0, stream>>>(wv, wvb, KVD * DIM);
  cvt_kernel<<<4096, 256, 0, stream>>>(wo, wob, DIM * DIM);

  // q/k/v projections (bf16 out), async-LDS double-buffered tiles
  gemm_tiled_kernel<<<(ROWS / BM) * (DIM / BN), 256, 0, stream>>>(
      xb, wqb, nullptr, qbuf, ROWS, DIM, DIM);
  gemm_tiled_kernel<<<(ROWS / BM) * (KVD / BN), 256, 0, stream>>>(
      xb, wkb, nullptr, kbuf, ROWS, KVD, DIM);
  gemm_tiled_kernel<<<(ROWS / BM) * (KVD / BN), 256, 0, stream>>>(
      xb, wvb, nullptr, vbuf, ROWS, KVD, DIM);

  lnrope_kernel<<<ROWS, 256, 0, stream>>>(qbuf, qw, qb, fc, DIM);
  lnrope_kernel<<<ROWS, 256, 0, stream>>>(kbuf, kw, kb, fc, KVD);
  vtrans_kernel<<<(ROWS * KVD + 255) / 256, 256, 0, stream>>>(vbuf, vtb);

  attn_kernel<<<(BB * NH * (SS / 16)) / 4, 128, 0, stream>>>(qbuf, kbuf, vtb, ab);

  // out = attn_out @ wo^T  (fp32 to d_out)
  gemm_tiled_kernel<<<(ROWS / BM) * (DIM / BN), 256, 0, stream>>>(
      ab, wob, (float*)d_out, nullptr, ROWS, DIM, DIM);
}
